// CAGNN_26096221291186
// MI455X (gfx1250) — matrix operand order repaired and checked by reading, not compiled, not run on hardware
//
#include <hip/hip_runtime.h>
#include <math.h>

#define NNODES 100000
#define NEDGES 1600000
#define D 64
#define NEG_SLOPE 0.2f

typedef __attribute__((ext_vector_type(16))) _Float16 v16h;
typedef __attribute__((ext_vector_type(8)))  float    v8f;

// Order-preserving float<->uint map (for atomicMax-based segment_max)
__device__ __forceinline__ unsigned enc_f32(float f) {
    unsigned u = __float_as_uint(f);
    return (u & 0x80000000u) ? ~u : (u | 0x80000000u);
}
__device__ __forceinline__ float dec_f32(unsigned u) {
    return __uint_as_float((u & 0x80000000u) ? (u ^ 0x80000000u) : ~u);
}
#define ENC_NEG_INF 0x007FFFFFu  // enc_f32(-inf)

// ---------------------------------------------------------------------------
// Init: zero the output accumulator, set emax = -inf, denom = 0
// ---------------------------------------------------------------------------
__global__ __launch_bounds__(256)
void gat_init(float* __restrict__ out, unsigned* __restrict__ emax,
              float* __restrict__ denom) {
    int i = blockIdx.x * blockDim.x + threadIdx.x;
    if (i < NNODES * D) out[i] = 0.0f;
    if (i < NNODES) { emax[i] = ENC_NEG_INF; denom[i] = 0.0f; }
}

// ---------------------------------------------------------------------------
// GEMM ft = feat @ W via WMMA f16->f32, fused with el/er attention reductions.
// Block = 128 threads (4 waves). Block handles 16 rows x 64 cols; wave w
// computes the 16x16 tile at columns [16w, 16w+16).
// ---------------------------------------------------------------------------
__global__ __launch_bounds__(128)
void gat_gemm_attn(const float* __restrict__ feat, const float* __restrict__ W,
                   const float* __restrict__ attn_l, const float* __restrict__ attn_r,
                   float* __restrict__ ft, float* __restrict__ el,
                   float* __restrict__ er) {
    __shared__ float sft[16][D + 1];

    const int lane    = threadIdx.x & 31;
    const int wave    = threadIdx.x >> 5;     // 0..3 -> column block
    const int rowBase = blockIdx.x * 16;      // NNODES % 16 == 0
    const int colBase = wave * 16;
    const int r = lane & 15;                  // A row / B column within tile
    const int h = lane >> 4;                  // lane-half selector

    // A fragments (16x32 f16): elems 0..15 = K(16h..16h+15),
    // per 32-wide K chunk (matches B order).
    v16h a0, a1;
    const float* arow = feat + (size_t)(rowBase + r) * D;
    #pragma unroll
    for (int j = 0; j < 8; ++j) {
        a0[j]     = (_Float16)arow[ 0 + 16 * h + j];
        a0[j + 8] = (_Float16)arow[ 8 + 16 * h + j];
        a1[j]     = (_Float16)arow[32 + 16 * h + j];
        a1[j + 8] = (_Float16)arow[40 + 16 * h + j];
    }
    // B fragments (32x16 f16): lane half h holds K = 16h+e, column = colBase+r
    v16h b0, b1;
    #pragma unroll
    for (int e = 0; e < 16; ++e) {
        b0[e] = (_Float16)W[( 0 + 16 * h + e) * D + colBase + r];
        b1[e] = (_Float16)W[(32 + 16 * h + e) * D + colBase + r];
    }

    v8f acc = {};
    acc = __builtin_amdgcn_wmma_f32_16x16x32_f16(false, a0, false, b0,
                                                 (short)0, acc, false, false);
    acc = __builtin_amdgcn_wmma_f32_16x16x32_f16(false, a1, false, b1,
                                                 (short)0, acc, false, false);

    // D layout: VGPR i -> row i + 8h, col = colBase + r
    #pragma unroll
    for (int i = 0; i < 8; ++i) {
        const int row = i + 8 * h;
        const int col = colBase + r;
        sft[row][col] = acc[i];
        ft[(size_t)(rowBase + row) * D + col] = acc[i];
    }
    __syncthreads();

    // el/er: per-row dot products against attn_l/attn_r (rows 0..15)
    if (threadIdx.x < 16) {
        float sl = 0.0f, sr = 0.0f;
        #pragma unroll 8
        for (int n = 0; n < D; ++n) {
            const float v = sft[threadIdx.x][n];
            sl += v * attn_l[n];
            sr += v * attn_r[n];
        }
        el[rowBase + threadIdx.x] = sl;
        er[rowBase + threadIdx.x] = sr;
    }
}

// ---------------------------------------------------------------------------
// Edge pass 1: leaky-relu score, segment-max into emax via uint atomicMax
// ---------------------------------------------------------------------------
__global__ __launch_bounds__(256)
void gat_edge_max(const int* __restrict__ src, const int* __restrict__ dst,
                  const float* __restrict__ el, const float* __restrict__ er,
                  unsigned* __restrict__ emax) {
    int i = blockIdx.x * blockDim.x + threadIdx.x;
    if (i >= NEDGES) return;
    float e = el[src[i]] + er[dst[i]];
    e = (e > 0.0f) ? e : NEG_SLOPE * e;
    atomicMax(emax + dst[i], enc_f32(e));
}

// ---------------------------------------------------------------------------
// Edge pass 2: w = exp(e - max[dst]); store per-edge weight, sum into denom
// ---------------------------------------------------------------------------
__global__ __launch_bounds__(256)
void gat_edge_exp(const int* __restrict__ src, const int* __restrict__ dst,
                  const float* __restrict__ el, const float* __restrict__ er,
                  const unsigned* __restrict__ emax, float* __restrict__ wbuf,
                  float* __restrict__ denom) {
    int i = blockIdx.x * blockDim.x + threadIdx.x;
    if (i >= NEDGES) return;
    const int d = dst[i];
    float e = el[src[i]] + er[d];
    e = (e > 0.0f) ? e : NEG_SLOPE * e;
    const float w = __expf(e - dec_f32(emax[d]));
    wbuf[i] = w;
    atomicAdd(denom + d, w);
}

// ---------------------------------------------------------------------------
// Edge pass 3: one wave32 per edge; scatter-add w * ft[src] into out[dst]
// (unnormalized; division by denom happens in finalize)
// ---------------------------------------------------------------------------
__global__ __launch_bounds__(256)
void gat_aggregate(const int* __restrict__ src, const int* __restrict__ dst,
                   const float* __restrict__ ft, const float* __restrict__ wbuf,
                   float* __restrict__ out) {
    const int eIdx = blockIdx.x * 8 + (threadIdx.x >> 5);
    if (eIdx >= NEDGES) return;
    const int lane = threadIdx.x & 31;
    const int s = src[eIdx];
    const int d = dst[eIdx];
    const float w = wbuf[eIdx];
    const float2 v = *(const float2*)(ft + (size_t)s * D + lane * 2);
    float* od = out + (size_t)d * D + lane * 2;
    atomicAdd(od,     v.x * w);
    atomicAdd(od + 1, v.y * w);
}

// ---------------------------------------------------------------------------
// Finalize: out = out/denom (0 for empty segments) + feat + bias
// ---------------------------------------------------------------------------
__global__ __launch_bounds__(256)
void gat_finalize(const float* __restrict__ feat, const float* __restrict__ bias,
                  const float* __restrict__ denom, float* __restrict__ out) {
    int i = blockIdx.x * blockDim.x + threadIdx.x;
    if (i >= NNODES * D) return;
    const int n = i >> 6;
    const int c = i & (D - 1);
    const float den = denom[n];
    const float r = (den > 0.0f) ? out[i] / den : 0.0f;
    out[i] = r + feat[i] + bias[c];
}

// ---------------------------------------------------------------------------
extern "C" void kernel_launch(void* const* d_in, const int* in_sizes, int n_in,
                              void* d_out, int out_size, void* d_ws, size_t ws_size,
                              hipStream_t stream) {
    (void)in_sizes; (void)n_in; (void)out_size; (void)ws_size;

    const float* feat   = (const float*)d_in[0];
    const float* W      = (const float*)d_in[1];
    const float* attn_l = (const float*)d_in[2];
    const float* attn_r = (const float*)d_in[3];
    const float* bias   = (const float*)d_in[4];
    const int*   src    = (const int*)d_in[5];
    const int*   dst    = (const int*)d_in[6];
    float* out = (float*)d_out;

    // Workspace layout (~34 MB total)
    float*    ft    = (float*)d_ws;                    // NNODES*D
    float*    el    = ft + (size_t)NNODES * D;         // NNODES
    float*    er    = el + NNODES;                     // NNODES
    unsigned* emax  = (unsigned*)(er + NNODES);        // NNODES
    float*    denom = (float*)(emax + NNODES);         // NNODES
    float*    wbuf  = denom + NNODES;                  // NEDGES

    gat_init<<<(NNODES * D + 255) / 256, 256, 0, stream>>>(out, emax, denom);
    gat_gemm_attn<<<NNODES / 16, 128, 0, stream>>>(feat, W, attn_l, attn_r,
                                                   ft, el, er);
    gat_edge_max<<<(NEDGES + 255) / 256, 256, 0, stream>>>(src, dst, el, er, emax);
    gat_edge_exp<<<(NEDGES + 255) / 256, 256, 0, stream>>>(src, dst, el, er,
                                                           emax, wbuf, denom);
    gat_aggregate<<<(NEDGES + 7) / 8, 256, 0, stream>>>(src, dst, ft, wbuf, out);
    gat_finalize<<<(NNODES * D + 255) / 256, 256, 0, stream>>>(feat, bias,
                                                               denom, out);
}
